// MemristorLinear_88046829568546
// MI455X (gfx1250) — compile-verified, matrix operand order checked by reading, not run on hardware
//
#include <hip/hip_runtime.h>

// ---- CDNA5 WMMA types ------------------------------------------------------
typedef __attribute__((ext_vector_type(16))) __bf16 v16bf;
typedef __attribute__((ext_vector_type(8)))  float  v8f;

#define TILE_M 128
#define TILE_N 128
#define NVALS_MAX 4096

// round-to-nearest-even fp32 -> bf16, packed pair into one dword
__device__ __forceinline__ unsigned int pack_bf16x2(float lo, float hi) {
    unsigned int ulo = __builtin_bit_cast(unsigned int, lo);
    unsigned int uhi = __builtin_bit_cast(unsigned int, hi);
    ulo = (ulo + 0x7FFFu + ((ulo >> 16) & 1u)) >> 16;
    uhi = (uhi + 0x7FFFu + ((uhi >> 16) & 1u));
    return (ulo & 0x0000FFFFu) | (uhi & 0xFFFF0000u);
}

// ============================================================================
// Phase 1: one-pass dequant + fp32->bf16 convert (memory-bound).
// ============================================================================
__global__ __launch_bounds__(256)
void dequant_convert(const float* __restrict__ x,
                     const float* __restrict__ values,
                     const int*   __restrict__ w_idx,
                     unsigned short* __restrict__ xb,
                     unsigned short* __restrict__ wb,
                     int total, int idx_mask)
{
    const int t = (blockIdx.x * 256 + threadIdx.x) * 8;
    if (t >= total) return;

    const float4 x0 = *reinterpret_cast<const float4*>(x + t);
    const float4 x1 = *reinterpret_cast<const float4*>(x + t + 4);
    uint4 xo;
    xo.x = pack_bf16x2(x0.x, x0.y);
    xo.y = pack_bf16x2(x0.z, x0.w);
    xo.z = pack_bf16x2(x1.x, x1.y);
    xo.w = pack_bf16x2(x1.z, x1.w);
    *reinterpret_cast<uint4*>(xb + t) = xo;

    const int4 i0 = *reinterpret_cast<const int4*>(w_idx + t);
    const int4 i1 = *reinterpret_cast<const int4*>(w_idx + t + 4);
    uint4 wo;
    wo.x = pack_bf16x2(values[i0.x & idx_mask], values[i0.y & idx_mask]);
    wo.y = pack_bf16x2(values[i0.z & idx_mask], values[i0.w & idx_mask]);
    wo.z = pack_bf16x2(values[i1.x & idx_mask], values[i1.y & idx_mask]);
    wo.w = pack_bf16x2(values[i1.z & idx_mask], values[i1.w & idx_mask]);
    *reinterpret_cast<uint4*>(wb + t) = wo;
}

// ============================================================================
// Phase 2: pure bf16 WMMA GEMM.  Block tile 128x128, K-step 64 (2 WMMA
// sub-steps), 8 waves in 2(M)x4(N), each wave 64x32 -> 16 WMMAs / iteration.
// K-prefetch double buffer held in NAMED SCALARS (qa0..qb3) so mem2reg keeps
// it register-resident -- arrays were being demoted to scratch.
// ============================================================================
#define TK2   64
#define LDS_H 80   // halfwords per row: 64 data + 16 pad -> 160 B (16B aligned)

__global__ __launch_bounds__(256)
__attribute__((amdgpu_waves_per_eu(1)))
void gemm_bf16_wmma(const unsigned short* __restrict__ xb,
                    const unsigned short* __restrict__ wb,
                    const float* __restrict__ bias,
                    float*       __restrict__ out,
                    int M, int N, int K)
{
    __shared__ __align__(16) unsigned short sA[TILE_M * LDS_H];
    __shared__ __align__(16) unsigned short sB[TILE_N * LDS_H];

    const int tid  = threadIdx.x;
    const int lane = tid & 31;
    const int wave = tid >> 5;

    const int m0 = blockIdx.y * TILE_M;
    const int n0 = blockIdx.x * TILE_N;

    // cooperative staging: tile 128 x 64 bf16 = 1024 x 16B / 256 thr = 4 each
    const int lrow = tid >> 3;   // 0..31
    const int lc8  = tid & 7;    // 16-byte column (8 halfs)

    const unsigned short* pA0 = xb + (size_t)(m0 + lrow) * K + lc8 * 8;
    const unsigned short* pA1 = pA0 + (size_t)32 * K;
    const unsigned short* pA2 = pA0 + (size_t)64 * K;
    const unsigned short* pA3 = pA0 + (size_t)96 * K;
    const unsigned short* pB0 = wb + (size_t)(n0 + lrow) * K + lc8 * 8;
    const unsigned short* pB1 = pB0 + (size_t)32 * K;
    const unsigned short* pB2 = pB0 + (size_t)64 * K;
    const unsigned short* pB3 = pB0 + (size_t)96 * K;

    // LDS destinations (constant per thread)
    unsigned short* dA0 = &sA[(lrow +  0) * LDS_H + lc8 * 8];
    unsigned short* dA1 = &sA[(lrow + 32) * LDS_H + lc8 * 8];
    unsigned short* dA2 = &sA[(lrow + 64) * LDS_H + lc8 * 8];
    unsigned short* dA3 = &sA[(lrow + 96) * LDS_H + lc8 * 8];
    unsigned short* dB0 = &sB[(lrow +  0) * LDS_H + lc8 * 8];
    unsigned short* dB1 = &sB[(lrow + 32) * LDS_H + lc8 * 8];
    unsigned short* dB2 = &sB[(lrow + 64) * LDS_H + lc8 * 8];
    unsigned short* dB3 = &sB[(lrow + 96) * LDS_H + lc8 * 8];

    // pipeline stage 0 in named scalar registers
    uint4 qa0 = *reinterpret_cast<const uint4*>(pA0);
    uint4 qa1 = *reinterpret_cast<const uint4*>(pA1);
    uint4 qa2 = *reinterpret_cast<const uint4*>(pA2);
    uint4 qa3 = *reinterpret_cast<const uint4*>(pA3);
    uint4 qb0 = *reinterpret_cast<const uint4*>(pB0);
    uint4 qb1 = *reinterpret_cast<const uint4*>(pB1);
    uint4 qb2 = *reinterpret_cast<const uint4*>(pB2);
    uint4 qb3 = *reinterpret_cast<const uint4*>(pB3);

    const int wm   = (wave & 1) * 64;
    const int wn   = (wave >> 1) * 32;
    const int lr   = lane & 15;
    const int hsel = (lane >> 4) & 1;

    v8f acc[4][2];
    const v8f zacc = {0.f, 0.f, 0.f, 0.f, 0.f, 0.f, 0.f, 0.f};
#pragma unroll
    for (int mt = 0; mt < 4; ++mt)
#pragma unroll
        for (int nt = 0; nt < 2; ++nt)
            acc[mt][nt] = zacc;

    for (int k0 = 0; k0 < K; k0 += TK2) {
        // regs -> LDS (no conversion: already bf16)
        *reinterpret_cast<uint4*>(dA0) = qa0;
        *reinterpret_cast<uint4*>(dA1) = qa1;
        *reinterpret_cast<uint4*>(dA2) = qa2;
        *reinterpret_cast<uint4*>(dA3) = qa3;
        *reinterpret_cast<uint4*>(dB0) = qb0;
        *reinterpret_cast<uint4*>(dB1) = qb1;
        *reinterpret_cast<uint4*>(dB2) = qb2;
        *reinterpret_cast<uint4*>(dB3) = qb3;
        __syncthreads();

        // prefetch next K-step into registers (overlaps the 16 WMMAs)
        if (k0 + TK2 < K) {
            const int kn = k0 + TK2;
            qa0 = *reinterpret_cast<const uint4*>(pA0 + kn);
            qa1 = *reinterpret_cast<const uint4*>(pA1 + kn);
            qa2 = *reinterpret_cast<const uint4*>(pA2 + kn);
            qa3 = *reinterpret_cast<const uint4*>(pA3 + kn);
            qb0 = *reinterpret_cast<const uint4*>(pB0 + kn);
            qb1 = *reinterpret_cast<const uint4*>(pB1 + kn);
            qb2 = *reinterpret_cast<const uint4*>(pB2 + kn);
            qb3 = *reinterpret_cast<const uint4*>(pB3 + kn);
            __builtin_prefetch(pA0 + (size_t)(k0 + 2 * TK2), 0, 0);
            __builtin_prefetch(pB0 + (size_t)(k0 + 2 * TK2), 0, 0);
        }

        // two 32-deep WMMA sub-steps
        union Frag { v16bf v; uint4 q[2]; };
#pragma unroll
        for (int kk = 0; kk < 2; ++kk) {
            const int koff = kk * 32;   // halfword offset within LDS row
            Frag a[4], b[2];
#pragma unroll
            for (int mt = 0; mt < 4; ++mt) {
                const int base = (wm + mt * 16 + lr) * LDS_H + koff + hsel * 8;
                a[mt].q[0] = *reinterpret_cast<const uint4*>(&sA[base]);
                a[mt].q[1] = *reinterpret_cast<const uint4*>(&sA[base + 16]);
            }
#pragma unroll
            for (int nt = 0; nt < 2; ++nt) {
                const int base = (wn + nt * 16 + lr) * LDS_H + koff + hsel * 8;
                b[nt].q[0] = *reinterpret_cast<const uint4*>(&sB[base]);
                b[nt].q[1] = *reinterpret_cast<const uint4*>(&sB[base + 16]);
            }
#pragma unroll
            for (int mt = 0; mt < 4; ++mt)
#pragma unroll
                for (int nt = 0; nt < 2; ++nt)
                    acc[mt][nt] = __builtin_amdgcn_wmma_f32_16x16x32_bf16(
                        false, a[mt].v, false, b[nt].v,
                        (short)0, acc[mt][nt], false, false);
        }
        __syncthreads();
    }

    // epilogue: C/D layout (VGPR r -> M=r / r+8 by lane half; N = lane&15)
#pragma unroll
    for (int nt = 0; nt < 2; ++nt) {
        const int col = n0 + wn + nt * 16 + lr;
        const float bv = bias[col];
#pragma unroll
        for (int mt = 0; mt < 4; ++mt) {
            const int rowb = m0 + wm + mt * 16 + hsel * 8;
#pragma unroll
            for (int r = 0; r < 8; ++r)
                out[(size_t)(rowb + r) * N + col] = acc[mt][nt][r] + bv;
        }
    }
}

// ============================================================================
// Fallback: fused dequant + WMMA GEMM (K-step 32, codebook in LDS).
// Used only if the workspace is too small for the two-phase path.
// ============================================================================
#define TK1 32
#define LDS_H1 40

__global__ __launch_bounds__(256)
__attribute__((amdgpu_waves_per_eu(1)))
void memristor_linear_fused(const float* __restrict__ x,
                            const float* __restrict__ values,
                            const int*   __restrict__ w_idx,
                            const float* __restrict__ bias,
                            float*       __restrict__ out,
                            int M, int N, int K, int n_vals)
{
    __shared__ __align__(16) float          s_vals[NVALS_MAX];
    __shared__ __align__(16) unsigned short sA[TILE_M * LDS_H1];
    __shared__ __align__(16) unsigned short sB[TILE_N * LDS_H1];

    const int tid  = threadIdx.x;
    const int lane = tid & 31;
    const int wave = tid >> 5;
    const int m0 = blockIdx.y * TILE_M;
    const int n0 = blockIdx.x * TILE_N;

    for (int i = tid; i < n_vals; i += 256) s_vals[i] = values[i];

    const int lrow = tid >> 3;
    const int lc4  = tid & 7;
    const float* xA0 = x + (size_t)(m0 + lrow) * K + lc4 * 4;
    const float* xA1 = xA0 + (size_t)32 * K;
    const float* xA2 = xA0 + (size_t)64 * K;
    const float* xA3 = xA0 + (size_t)96 * K;
    const int* wB0 = w_idx + (size_t)(n0 + lrow) * K + lc4 * 4;
    const int* wB1 = wB0 + (size_t)32 * K;
    const int* wB2 = wB0 + (size_t)64 * K;
    const int* wB3 = wB0 + (size_t)96 * K;

    unsigned short* dA0 = &sA[(lrow +  0) * LDS_H1 + lc4 * 4];
    unsigned short* dA1 = &sA[(lrow + 32) * LDS_H1 + lc4 * 4];
    unsigned short* dA2 = &sA[(lrow + 64) * LDS_H1 + lc4 * 4];
    unsigned short* dA3 = &sA[(lrow + 96) * LDS_H1 + lc4 * 4];
    unsigned short* dB0 = &sB[(lrow +  0) * LDS_H1 + lc4 * 4];
    unsigned short* dB1 = &sB[(lrow + 32) * LDS_H1 + lc4 * 4];
    unsigned short* dB2 = &sB[(lrow + 64) * LDS_H1 + lc4 * 4];
    unsigned short* dB3 = &sB[(lrow + 96) * LDS_H1 + lc4 * 4];

    float4 ra0 = *reinterpret_cast<const float4*>(xA0);
    float4 ra1 = *reinterpret_cast<const float4*>(xA1);
    float4 ra2 = *reinterpret_cast<const float4*>(xA2);
    float4 ra3 = *reinterpret_cast<const float4*>(xA3);
    int4   rb0 = *reinterpret_cast<const int4*>(wB0);
    int4   rb1 = *reinterpret_cast<const int4*>(wB1);
    int4   rb2 = *reinterpret_cast<const int4*>(wB2);
    int4   rb3 = *reinterpret_cast<const int4*>(wB3);
    __syncthreads();

    const int wm   = (wave & 1) * 64;
    const int wn   = (wave >> 1) * 32;
    const int lr   = lane & 15;
    const int hsel = (lane >> 4) & 1;

    v8f acc[4][2];
    const v8f zacc = {0.f, 0.f, 0.f, 0.f, 0.f, 0.f, 0.f, 0.f};
#pragma unroll
    for (int mt = 0; mt < 4; ++mt)
#pragma unroll
        for (int nt = 0; nt < 2; ++nt) acc[mt][nt] = zacc;

    const int idx_mask = n_vals - 1;

    for (int k0 = 0; k0 < K; k0 += TK1) {
        *reinterpret_cast<uint2*>(dA0) =
            make_uint2(pack_bf16x2(ra0.x, ra0.y), pack_bf16x2(ra0.z, ra0.w));
        *reinterpret_cast<uint2*>(dA1) =
            make_uint2(pack_bf16x2(ra1.x, ra1.y), pack_bf16x2(ra1.z, ra1.w));
        *reinterpret_cast<uint2*>(dA2) =
            make_uint2(pack_bf16x2(ra2.x, ra2.y), pack_bf16x2(ra2.z, ra2.w));
        *reinterpret_cast<uint2*>(dA3) =
            make_uint2(pack_bf16x2(ra3.x, ra3.y), pack_bf16x2(ra3.z, ra3.w));
        *reinterpret_cast<uint2*>(dB0) =
            make_uint2(pack_bf16x2(s_vals[rb0.x & idx_mask], s_vals[rb0.y & idx_mask]),
                       pack_bf16x2(s_vals[rb0.z & idx_mask], s_vals[rb0.w & idx_mask]));
        *reinterpret_cast<uint2*>(dB1) =
            make_uint2(pack_bf16x2(s_vals[rb1.x & idx_mask], s_vals[rb1.y & idx_mask]),
                       pack_bf16x2(s_vals[rb1.z & idx_mask], s_vals[rb1.w & idx_mask]));
        *reinterpret_cast<uint2*>(dB2) =
            make_uint2(pack_bf16x2(s_vals[rb2.x & idx_mask], s_vals[rb2.y & idx_mask]),
                       pack_bf16x2(s_vals[rb2.z & idx_mask], s_vals[rb2.w & idx_mask]));
        *reinterpret_cast<uint2*>(dB3) =
            make_uint2(pack_bf16x2(s_vals[rb3.x & idx_mask], s_vals[rb3.y & idx_mask]),
                       pack_bf16x2(s_vals[rb3.z & idx_mask], s_vals[rb3.w & idx_mask]));
        __syncthreads();

        if (k0 + TK1 < K) {
            const int kn = k0 + TK1;
            ra0 = *reinterpret_cast<const float4*>(xA0 + kn);
            ra1 = *reinterpret_cast<const float4*>(xA1 + kn);
            ra2 = *reinterpret_cast<const float4*>(xA2 + kn);
            ra3 = *reinterpret_cast<const float4*>(xA3 + kn);
            rb0 = *reinterpret_cast<const int4*>(wB0 + kn);
            rb1 = *reinterpret_cast<const int4*>(wB1 + kn);
            rb2 = *reinterpret_cast<const int4*>(wB2 + kn);
            rb3 = *reinterpret_cast<const int4*>(wB3 + kn);
        }

        union Frag { v16bf v; uint4 q[2]; };
        Frag a[4], b[2];
#pragma unroll
        for (int mt = 0; mt < 4; ++mt) {
            const int base = (wm + mt * 16 + lr) * LDS_H1 + hsel * 8;
            a[mt].q[0] = *reinterpret_cast<const uint4*>(&sA[base]);
            a[mt].q[1] = *reinterpret_cast<const uint4*>(&sA[base + 16]);
        }
#pragma unroll
        for (int nt = 0; nt < 2; ++nt) {
            const int base = (wn + nt * 16 + lr) * LDS_H1 + hsel * 8;
            b[nt].q[0] = *reinterpret_cast<const uint4*>(&sB[base]);
            b[nt].q[1] = *reinterpret_cast<const uint4*>(&sB[base + 16]);
        }
#pragma unroll
        for (int mt = 0; mt < 4; ++mt)
#pragma unroll
            for (int nt = 0; nt < 2; ++nt)
                acc[mt][nt] = __builtin_amdgcn_wmma_f32_16x16x32_bf16(
                    false, a[mt].v, false, b[nt].v,
                    (short)0, acc[mt][nt], false, false);
        __syncthreads();
    }

#pragma unroll
    for (int nt = 0; nt < 2; ++nt) {
        const int col = n0 + wn + nt * 16 + lr;
        const float bv = bias[col];
#pragma unroll
        for (int mt = 0; mt < 4; ++mt) {
            const int rowb = m0 + wm + mt * 16 + hsel * 8;
#pragma unroll
            for (int r = 0; r < 8; ++r)
                out[(size_t)(rowb + r) * N + col] = acc[mt][nt][r] + bv;
        }
    }
}

// ============================================================================
extern "C" void kernel_launch(void* const* d_in, const int* in_sizes, int n_in,
                              void* d_out, int out_size, void* d_ws, size_t ws_size,
                              hipStream_t stream) {
    const float* x      = (const float*)d_in[0];
    const float* values = (const float*)d_in[1];
    const int*   w_idx  = (const int*)  d_in[2];
    const float* bias   = (const float*)d_in[3];
    float*       out    = (float*)d_out;

    const int n_vals = in_sizes[1];          // 4096
    const int N      = in_sizes[3];          // OUT_F
    const int K      = in_sizes[2] / N;      // IN_F
    const int Mtok   = in_sizes[0] / K;      // tokens

    const size_t xb_elems = (size_t)Mtok * K;
    const size_t wb_elems = (size_t)N * K;
    const size_t need     = (xb_elems + wb_elems) * sizeof(unsigned short);

    dim3 grid(N / TILE_N, Mtok / TILE_M);
    dim3 block(256);

    if (ws_size >= need) {
        unsigned short* xb = (unsigned short*)d_ws;
        unsigned short* wb = xb + xb_elems;

        const int total  = (int)xb_elems;           // == wb_elems here
        const int blocks = (total / 8 + 255) / 256;
        dequant_convert<<<blocks, block, 0, stream>>>(
            x, values, w_idx, xb, wb, total, n_vals - 1);

        gemm_bf16_wmma<<<grid, block, 0, stream>>>(
            xb, wb, bias, out, Mtok, N, K);
    } else {
        memristor_linear_fused<<<grid, block, 0, stream>>>(
            x, values, w_idx, bias, out, Mtok, N, K, n_vals);
    }
}
